// CausalTransformer_63462436765808
// MI455X (gfx1250) — compile-verified, matrix operand order checked by reading, not run on hardware
//
#include <hip/hip_runtime.h>
#include <math.h>

// Problem dims (fixed by the reference)
#define T_TOK 4096   // B*N tokens
#define D_DIM 1024
#define H_DIM 4096
#define E_NUM 8
#define HCHUNK 128   // H processed per outer iteration (16 cols per wave)
#define MTILE 32     // tokens per block (two 16-row WMMA M-tiles)

typedef __attribute__((ext_vector_type(16))) __bf16 v16bf;
typedef __attribute__((ext_vector_type(8)))  float  v8f;

// ---- bf16 helpers (manual RNE pack; cvt_pk builtin absent on this clang) --
__device__ __forceinline__ unsigned int cvt2_bf16(float a, float b) {
  unsigned int ua = __float_as_uint(a);
  unsigned int ub = __float_as_uint(b);
  ua += 0x7fffu + ((ua >> 16) & 1u);                   // RNE
  ub += 0x7fffu + ((ub >> 16) & 1u);
  return (ua >> 16) | (ub & 0xffff0000u);              // lo = a, hi = b
}
__device__ __forceinline__ unsigned short cvt1_bf16(float a) {
  unsigned int ua = __float_as_uint(a);
  ua += 0x7fffu + ((ua >> 16) & 1u);
  return (unsigned short)(ua >> 16);
}

// Branch-free fast GELU (tanh form): v_exp_f32 + v_rcp_f32, no EXEC churn.
__device__ __forceinline__ float gelu_f(float v) {
  const float u  = 0.7978845608028654f * (v + 0.044715f * v * v * v);
  const float e  = __expf(-2.0f * fabsf(u));
  const float th = __builtin_copysignf((1.0f - e) * __builtin_amdgcn_rcpf(1.0f + e), u);
  return 0.5f * v * (1.0f + th);
}

// ---- WMMA operand loaders ----------------------------------------------
// 16-bit A (16xK tile, one 32-wide K-slice) per ISA 7.12.2:
//   lanes 0-15 : row M=lane,    K = {k0..k0+7, k0+16..k0+23}
//   lanes 16-31: row M=lane-16, K = {k0+8..k0+15, k0+24..k0+31}
// B (32x16) uses the same lane/half striping with "row" = column N.
__device__ __forceinline__ v16bf lds_operand(const unsigned short* base,
                                             int row_stride, int k0) {
  const int lane = threadIdx.x & 31;
  const int half = lane >> 4;
  const int r    = lane & 15;
  const unsigned short* p = base + r * row_stride + k0 + half * 8;
  union { v16bf v; uint4 q[2]; } u;
  u.q[0] = *(const uint4*)(p);
  u.q[1] = *(const uint4*)(p + 16);
  return u.v;
}

// Packed-weight B-operand tile: 1 KB per 32(K)x16(N) tile.
// Plane 0 (512B): all 32 lanes' VGPRs 0..3; plane 1 (512B): VGPRs 4..7.
__device__ __forceinline__ v16bf load_b_packed(const unsigned short* __restrict__ mat,
                                               int KT, int nt, int kt) {
  const int lane = threadIdx.x & 31;
  const unsigned short* t = mat + ((size_t)(nt * KT + kt) << 9);  // 512 elems/tile
  union { v16bf v; uint4 q[2]; } u;
  u.q[0] = *(const uint4*)(t + lane * 8);
  u.q[1] = *(const uint4*)(t + 256 + lane * 8);
  return u.v;
}

// Direct-gather fallback: B from row-major fp32 W[k][n], wbase[k*ld + n].
__device__ __forceinline__ v16bf gather_b_f32(const float* __restrict__ wbase,
                                              int ld, int n0, int k0) {
  const int lane = threadIdx.x & 31;
  const int half = lane >> 4;
  const float* p = wbase + (size_t)(k0 + half * 8) * ld + n0 + (lane & 15);
  union { v16bf v; unsigned int u[8]; } r;
#pragma unroll
  for (int i = 0; i < 4; ++i)
    r.u[i] = cvt2_bf16(p[(2 * i) * ld], p[(2 * i + 1) * ld]);
  const float* p2 = p + (size_t)16 * ld;
#pragma unroll
  for (int i = 0; i < 4; ++i)
    r.u[4 + i] = cvt2_bf16(p2[(2 * i) * ld], p2[(2 * i + 1) * ld]);
  return r.v;
}

// ---- kernel 0: zero the tiny workspace counters -------------------------
__global__ void init_kernel(int* __restrict__ counts, float* __restrict__ imp) {
  const int t = threadIdx.x;
  if (t < E_NUM) { counts[t] = 0; imp[t] = 0.0f; }
}

// ---- kernel P: pre-pack fp32 weights -> swizzled bf16 B-operand tiles ----
__global__ __launch_bounds__(256) void prepack_kernel(
    const float* __restrict__ src, unsigned short* __restrict__ dst,
    int K, int N) {
  const int KT = K >> 5, NT = N >> 4;
  const int tilesPerMat = KT * NT;
  const int wgid = blockIdx.x * 8 + ((int)threadIdx.x >> 5);
  const int lane = threadIdx.x & 31;
  const int e  = wgid / tilesPerMat;
  const int r  = wgid - e * tilesPerMat;
  const int nt = r / KT, kt = r - nt * KT;

  const float* s = src + (size_t)e * K * N
                 + (size_t)(kt * 32 + (lane >> 4) * 8) * N + nt * 16 + (lane & 15);
  unsigned int u[8];
#pragma unroll
  for (int i = 0; i < 4; ++i)
    u[i] = cvt2_bf16(s[(size_t)(2 * i) * N], s[(size_t)(2 * i + 1) * N]);
  const float* s2 = s + (size_t)16 * N;
#pragma unroll
  for (int i = 0; i < 4; ++i)
    u[4 + i] = cvt2_bf16(s2[(size_t)(2 * i) * N], s2[(size_t)(2 * i + 1) * N]);

  unsigned short* t = dst + ((size_t)wgid << 9);
  *(uint4*)(t + lane * 8)       = make_uint4(u[0], u[1], u[2], u[3]);
  *(uint4*)(t + 256 + lane * 8) = make_uint4(u[4], u[5], u[6], u[7]);
}

// ---- kernel 1: router (logits, gumbel-argmax routing, importance) -------
__global__ __launch_bounds__(256) void router_kernel(
    const float* __restrict__ x, const float* __restrict__ gumbel,
    const float* __restrict__ rw, const float* __restrict__ rb,
    int* __restrict__ counts, int* __restrict__ lists,
    float* __restrict__ imp) {
  __shared__ float s_imp[E_NUM];
  const int tid  = threadIdx.x;
  const int wv   = tid >> 5;
  const int lane = tid & 31;
  if (tid < E_NUM) s_imp[tid] = 0.0f;
  __syncthreads();

  const int t = blockIdx.x * 8 + wv;
  float acc[E_NUM] = {0.f, 0.f, 0.f, 0.f, 0.f, 0.f, 0.f, 0.f};
  const float* xr = x + (size_t)t * D_DIM;
  for (int j = lane; j < D_DIM; j += 32) {
    const float xv = xr[j];
    const float4* rwj = (const float4*)(rw + (size_t)j * E_NUM);
    const float4 r0 = rwj[0], r1 = rwj[1];
    acc[0] += xv * r0.x; acc[1] += xv * r0.y; acc[2] += xv * r0.z; acc[3] += xv * r0.w;
    acc[4] += xv * r1.x; acc[5] += xv * r1.y; acc[6] += xv * r1.z; acc[7] += xv * r1.w;
  }
#pragma unroll
  for (int e = 0; e < E_NUM; ++e)
#pragma unroll
    for (int off = 16; off; off >>= 1)
      acc[e] += __shfl_xor(acc[e], off, 32);

  if (lane == 0) {
    float logit[E_NUM], lg[E_NUM];
    float m1 = -3.4e38f;
#pragma unroll
    for (int e = 0; e < E_NUM; ++e) {
      logit[e] = acc[e] + rb[e];
      lg[e]    = logit[e] + gumbel[(size_t)t * E_NUM + e];
      m1 = fmaxf(m1, logit[e]);
    }
    float s1 = 0.0f;
#pragma unroll
    for (int e = 0; e < E_NUM; ++e) s1 += __expf(logit[e] - m1);
    const float inv_s1 = __builtin_amdgcn_rcpf(s1);
    int arg = 0; float best = lg[0];
#pragma unroll
    for (int e = 1; e < E_NUM; ++e)
      if (lg[e] > best) { best = lg[e]; arg = e; }   // first-max, like argmax
#pragma unroll
    for (int e = 0; e < E_NUM; ++e)
      atomicAdd(&s_imp[e], __expf(logit[e] - m1) * inv_s1);
    const int pos = atomicAdd(&counts[arg], 1);
    lists[arg * T_TOK + pos] = t;
  }
  __syncthreads();
  if (tid < E_NUM) atomicAdd(&imp[tid], s_imp[tid]);
}

// ---- kernel 2: aux loss -------------------------------------------------
__global__ void aux_kernel(const float* __restrict__ imp, float* __restrict__ out_aux) {
  const int l = threadIdx.x;
  float v = 0.0f;
  if (l < E_NUM) {
    const float d = imp[l] * (1.0f / (float)T_TOK) - (1.0f / (float)E_NUM);
    v = d * d;
  }
#pragma unroll
  for (int off = 16; off; off >>= 1) v += __shfl_xor(v, off, 32);
  if (l == 0) *out_aux = v * (1.0f / (float)E_NUM);
}

// ---- kernel 3: routed fused FFN (WMMA workhorse) ------------------------
// grid.x = E_NUM * (T_TOK/MTILE); block = 256 threads = 8 waves.
// One block = one 32-token tile of one expert, all 1024 output columns.
// B operands flow through a 4-deep register ring: 4 load-pairs are always
// in flight (structurally — all ring slots live), so WMMAs wait on
// loadcnt<=6 instead of 0.
template <bool PACKED>
__global__ __launch_bounds__(256) void moe_ffn_kernel(
    const float* __restrict__ x,  const float* __restrict__ w1,
    const float* __restrict__ b1, const float* __restrict__ w2,
    const float* __restrict__ b2, const int* __restrict__ counts,
    const int* __restrict__ lists,
    const unsigned short* __restrict__ w1p,
    const unsigned short* __restrict__ w2p,
    float* __restrict__ out) {
  const int e    = blockIdx.x >> 7;               // / 128 tiles per expert
  const int tile = blockIdx.x & 127;
  const int cnt  = counts[e];
  if (tile * MTILE >= cnt) return;                // idle tile: exit fast
  const int rows = min(MTILE, cnt - tile * MTILE);

  __shared__ alignas(16) unsigned short lds_x[MTILE * 1032]; // bf16 x tile (~66 KB)
  __shared__ alignas(16) unsigned short lds_h[MTILE * 136];  // bf16 hidden chunk
  __shared__ int s_tok[MTILE];

  const int tid  = threadIdx.x;
  const int wv   = tid >> 5;
  const int lane = tid & 31;
  const int half = lane >> 4;
  const int l16  = lane & 15;

  if (tid < MTILE) s_tok[tid] = lists[e * T_TOK + tile * MTILE + min(tid, rows - 1)];
  __syncthreads();

  // Stage x tile (32 x 1024 fp32 -> bf16 LDS, [m][k], stride 1032: row pitch
  // 2064B = 256B*8 + 16B, so the 16 row-parallel ds_load_b128 are conflict-free)
  {
    const int r  = tid >> 3;                      // 8 threads per row
    const int c0 = (tid & 7) * 128;
    const float* xr = x + (size_t)s_tok[r] * D_DIM + c0;
    unsigned short* dst = lds_x + r * 1032 + c0;
#pragma unroll
    for (int c = 0; c < 128; c += 2) {
      const float2 f = *(const float2*)(xr + c);
      *(unsigned int*)(dst + c) = cvt2_bf16(f.x, f.y);
    }
  }
  __syncthreads();

  const float* w1e = w1 + (size_t)e * D_DIM * H_DIM;
  const float* w2e = w2 + (size_t)e * H_DIM * D_DIM;
  const unsigned short* w1pe = w1p + (size_t)e * (D_DIM / 32) * (H_DIM / 16) * 512;
  const unsigned short* w2pe = w2p + (size_t)e * (H_DIM / 32) * (D_DIM / 16) * 512;

  const v8f vzero = {0.f, 0.f, 0.f, 0.f, 0.f, 0.f, 0.f, 0.f};
  v8f acc0[8], acc1[8];                           // 2 M-tiles x 8 N-tiles
#pragma unroll
  for (int nt = 0; nt < 8; ++nt) { acc0[nt] = vzero; acc1[nt] = vzero; }

  for (int h0 = 0; h0 < H_DIM; h0 += HCHUNK) {
    const int n0  = h0 + wv * 16;                 // this wave's hidden columns
    const int nt1 = (h0 >> 4) + wv;               // packed N-tile index (GEMM1)

    auto load_b1 = [&](int k0) -> v16bf {
      return PACKED ? load_b_packed(w1pe, D_DIM / 32, nt1, k0 >> 5)
                    : gather_b_f32(w1e, H_DIM, n0, k0);
    };
    auto load_b2 = [&](int nt, int kabs) -> v16bf {
      return PACKED ? load_b_packed(w2pe, H_DIM / 32, wv * 8 + nt, kabs >> 5)
                    : gather_b_f32(w2e, D_DIM, wv * 128 + nt * 16, kabs);
    };

    // ---- GEMM1 (4-deep B ring, fully unrolled): K = D = 1024, 32 steps ----
    v8f h0acc = vzero, h1acc = vzero;
    v16bf b1r[4];
#pragma unroll
    for (int j = 0; j < 4; ++j) b1r[j] = load_b1(j * 32);
    v16bf a0cur = lds_operand(lds_x,             1032, 0);
    v16bf a1cur = lds_operand(lds_x + 16 * 1032, 1032, 0);
#pragma unroll
    for (int i = 0; i < 32; ++i) {
      v16bf a0nxt, a1nxt;
      if (i < 31) {
        a0nxt = lds_operand(lds_x,             1032, (i + 1) * 32);
        a1nxt = lds_operand(lds_x + 16 * 1032, 1032, (i + 1) * 32);
      }
      const v16bf bb = b1r[i & 3];
      h0acc = __builtin_amdgcn_wmma_f32_16x16x32_bf16(false, a0cur, false, bb,
                                                      (short)0, h0acc, false, false);
      h1acc = __builtin_amdgcn_wmma_f32_16x16x32_bf16(false, a1cur, false, bb,
                                                      (short)0, h1acc, false, false);
      if (i + 4 < 32) b1r[i & 3] = load_b1((i + 4) * 32);
      if (i < 31) { a0cur = a0nxt; a1cur = a1nxt; }
    }

    // GEMM2 ring prologue issued early: overlaps the GELU VALU + barrier.
    v16bf b2r[4];
#pragma unroll
    for (int j = 0; j < 4; ++j) b2r[j] = load_b2(j, h0);

    // bias + fast GELU, store bf16 hidden chunk
    const float bias1 = b1[e * H_DIM + n0 + l16];
#pragma unroll
    for (int r = 0; r < 8; ++r) {
      const int m = r + half * 8;
      lds_h[m * 136 + wv * 16 + l16]        = cvt1_bf16(gelu_f(h0acc[r] + bias1));
      lds_h[(m + 16) * 136 + wv * 16 + l16] = cvt1_bf16(gelu_f(h1acc[r] + bias1));
    }
    __syncthreads();

    if (lane == 0 && h0 + HCHUNK < H_DIM)         // warm L2 for next chunk
      __builtin_prefetch(
          PACKED ? (const void*)(w1pe + ((size_t)(((h0 + HCHUNK) >> 4) + wv) * (D_DIM / 32) << 9))
                 : (const void*)(w1e + (size_t)(h0 + HCHUNK + wv * 16)),
          0, 1);

    // ---- GEMM2 (4-deep B ring, fully unrolled): 32 (k0,nt) steps ----
    v16bf a0, a1;
#pragma unroll
    for (int i = 0; i < 32; ++i) {                // i = k0idx*8 + nt
      const int k0 = (i >> 3) * 32;
      const int nt = i & 7;
      if (nt == 0) {
        a0 = lds_operand(lds_h,            136, k0);
        a1 = lds_operand(lds_h + 16 * 136, 136, k0);
      }
      const v16bf bb = b2r[i & 3];
      acc0[nt] = __builtin_amdgcn_wmma_f32_16x16x32_bf16(false, a0, false, bb,
                                                         (short)0, acc0[nt], false, false);
      acc1[nt] = __builtin_amdgcn_wmma_f32_16x16x32_bf16(false, a1, false, bb,
                                                         (short)0, acc1[nt], false, false);
      if (i + 4 < 32) {
        const int ni = i + 4;
        b2r[i & 3] = load_b2(ni & 7, h0 + (ni >> 3) * 32);
      }
    }
    __syncthreads();                              // lds_h reused next iteration
  }

  // Epilogue: +b2, scatter rows to their tokens (one-hot combine == direct store)
#pragma unroll
  for (int nt = 0; nt < 8; ++nt) {
    const int col   = wv * 128 + nt * 16 + l16;
    const float b2v = b2[e * D_DIM + col];
#pragma unroll
    for (int r = 0; r < 8; ++r) {
      const int m = r + half * 8;                 // C layout: M = r + 8*lane_half
      if (m < rows)
        out[(size_t)s_tok[m] * D_DIM + col] = acc0[nt][r] + b2v;
      if (m + 16 < rows)
        out[(size_t)s_tok[m + 16] * D_DIM + col] = acc1[nt][r] + b2v;
    }
  }
}

// ---- host launcher ------------------------------------------------------
extern "C" void kernel_launch(void* const* d_in, const int* in_sizes, int n_in,
                              void* d_out, int out_size, void* d_ws, size_t ws_size,
                              hipStream_t stream) {
  const float* x      = (const float*)d_in[0];
  const float* gumbel = (const float*)d_in[1];
  const float* rw     = (const float*)d_in[2];
  const float* rb     = (const float*)d_in[3];
  const float* w1     = (const float*)d_in[4];
  const float* b1     = (const float*)d_in[5];
  const float* w2     = (const float*)d_in[6];
  const float* b2     = (const float*)d_in[7];

  float* out = (float*)d_out;                        // (B,N,D) flat
  float* aux = out + (size_t)T_TOK * D_DIM;          // trailing scalar

  int*   counts = (int*)d_ws;
  float* imp    = (float*)((char*)d_ws + 32);
  int*   lists  = (int*)((char*)d_ws + 64);
  const size_t W1P_OFS = 256 * 1024;
  const size_t WMAT_SZ = (size_t)E_NUM * D_DIM * H_DIM * 2;   // 64 MB each
  const size_t WS_NEEDED = W1P_OFS + 2 * WMAT_SZ;
  unsigned short* w1p = (unsigned short*)((char*)d_ws + W1P_OFS);
  unsigned short* w2p = (unsigned short*)((char*)d_ws + W1P_OFS + WMAT_SZ);
  const bool packed = ws_size >= WS_NEEDED;

  init_kernel<<<1, 32, 0, stream>>>(counts, imp);
  router_kernel<<<T_TOK / 8, 256, 0, stream>>>(x, gumbel, rw, rb, counts, lists, imp);
  aux_kernel<<<1, 32, 0, stream>>>(imp, aux);

  if (packed) {
    prepack_kernel<<<(E_NUM * (D_DIM / 32) * (H_DIM / 16)) / 8, 256, 0, stream>>>(
        w1, w1p, D_DIM, H_DIM);
    prepack_kernel<<<(E_NUM * (H_DIM / 32) * (D_DIM / 16)) / 8, 256, 0, stream>>>(
        w2, w2p, H_DIM, D_DIM);
    moe_ffn_kernel<true><<<E_NUM * (T_TOK / MTILE), 256, 0, stream>>>(
        x, w1, b1, w2, b2, counts, lists, w1p, w2p, out);
  } else {
    moe_ffn_kernel<false><<<E_NUM * (T_TOK / MTILE), 256, 0, stream>>>(
        x, w1, b1, w2, b2, counts, lists, w1p, w2p, out);
  }
}
